// TransformerBlock_49254684951028
// MI455X (gfx1250) — matrix operand order repaired and checked by reading, not compiled, not run on hardware
//
#include <hip/hip_runtime.h>

// ---------------------------------------------------------------------------
// Types for CDNA5 WMMA (wave32, 16x16x32 bf16 -> f32)
// ---------------------------------------------------------------------------
typedef __attribute__((ext_vector_type(16))) __bf16 v16bf;
typedef __attribute__((ext_vector_type(8)))  float  v8f;

struct B32B { uint4 lo; uint4 hi; };   // 32-byte carrier for a 16 x bf16 fragment

static __device__ __forceinline__ unsigned short f2bf(float f) {
  unsigned int u = __builtin_bit_cast(unsigned int, f);
  unsigned int r = u + 0x7FFFu + ((u >> 16) & 1u);   // round-to-nearest-even
  return (unsigned short)(r >> 16);
}

// 16 contiguous bf16 (B-matrix fragment: lane = column N, 16 consecutive K)
static __device__ __forceinline__ v16bf ld_b_frag(const unsigned short* p) {
  B32B t;
  t.lo = *(const uint4*)(p);
  t.hi = *(const uint4*)(p + 8);
  return __builtin_bit_cast(v16bf, t);
}

// A-matrix fragment: elems 0..15 = K[half*16 .. half*16+15]
// (contiguous, matches B fragments; row pointer already at k-tile)
static __device__ __forceinline__ v16bf ld_a_frag(const unsigned short* row, int half) {
  B32B t;
  t.lo = *(const uint4*)(row + half * 16);
  t.hi = *(const uint4*)(row + half * 16 + 8);
  return __builtin_bit_cast(v16bf, t);
}

static __device__ __forceinline__ v8f wmma_bf16(v16bf a, v16bf b, v8f c) {
  return __builtin_amdgcn_wmma_f32_16x16x32_bf16(false, a, false, b, (short)0, c,
                                                 false, false);
}

static __device__ __forceinline__ float gelu_new(float x) {
  return 0.5f * x * (1.0f + tanhf(0.7978845608028654f * (x + 0.044715f * x * x * x)));
}

// Async global->LDS copy of 16 bytes (CDNA5 GLOBAL_LOAD_ASYNC_TO_LDS_B128,
// tracked with ASYNCcnt).  ldsOff is the byte offset in LDS (low 32 bits of a
// generic pointer to __shared__), gaddr the 64-bit global address.
static __device__ __forceinline__ void async_cp16(unsigned ldsOff, const void* gaddr) {
  asm volatile("global_load_async_to_lds_b128 %0, %1, off"
               :: "v"(ldsOff), "v"((unsigned long long)(uintptr_t)gaddr)
               : "memory");
}

// ---------------------------------------------------------------------------
// Weight packing: fp32 -> bf16, transposed so B-fragments are K-contiguous
// ---------------------------------------------------------------------------

// W_{Q,K,V}: [H=12, D=768, Dh=64] fp32  ->  Wt[n=h*64+e][k=d] bf16  (768x768)
__global__ void pack_qkv_w(const float* __restrict__ W, unsigned short* __restrict__ Wt) {
  int idx = blockIdx.x * 256 + threadIdx.x;            // over 768*768
  int n = idx / 768, k = idx - n * 768;
  int h = n >> 6, e = n & 63;
  Wt[idx] = f2bf(W[(h * 768 + k) * 64 + e]);
}

// generic transpose-pack: src[R][C] fp32 -> dst[C][R] bf16
__global__ void transpose_pack(const float* __restrict__ src, unsigned short* __restrict__ dst,
                               int R, int C) {
  int idx = blockIdx.x * 256 + threadIdx.x;
  if (idx >= R * C) return;
  int r = idx / C, c = idx - r * C;
  dst[(size_t)c * R + r] = f2bf(src[idx]);
}

// ---------------------------------------------------------------------------
// LayerNorm over D=768: one 256-thread block per row, fp32 in -> bf16 out
// ---------------------------------------------------------------------------
__global__ void ln_kernel(const float* __restrict__ x, const float* __restrict__ w,
                          const float* __restrict__ b, unsigned short* __restrict__ out) {
  int row = blockIdx.x;
  int t   = threadIdx.x;
  const float* xr = x + (size_t)row * 768;
  float v0 = xr[t], v1 = xr[t + 256], v2 = xr[t + 512];
  __shared__ float rs[256], rss[256];
  rs[t]  = v0 + v1 + v2;
  rss[t] = v0 * v0 + v1 * v1 + v2 * v2;
  __syncthreads();
  for (int o = 128; o > 0; o >>= 1) {
    if (t < o) { rs[t] += rs[t + o]; rss[t] += rss[t + o]; }
    __syncthreads();
  }
  float mean = rs[0] * (1.0f / 768.0f);
  float var  = rss[0] * (1.0f / 768.0f) - mean * mean;
  float rstd = rsqrtf(var + 1e-5f);
  unsigned short* o0 = out + (size_t)row * 768;
  o0[t]       = f2bf((v0 - mean) * rstd * w[t]       + b[t]);
  o0[t + 256] = f2bf((v1 - mean) * rstd * w[t + 256] + b[t + 256]);
  o0[t + 512] = f2bf((v2 - mean) * rstd * w[t + 512] + b[t + 512]);
}

// ---------------------------------------------------------------------------
// bf16 WMMA GEMM with async-LDS double buffering:
//   C[M,N] = A[M,K] * Bt[N,K]^T (+bias), epilogues:
//   MODE 0: out bf16 in [B,H,S,Dh] head layout      (Q, K)
//   MODE 1: out bf16 in [B,H,Dh,S] transposed       (V)
//   MODE 2: outF fp32 = resid + acc + bias          (O-proj, MLP-out)
//   MODE 3: out bf16 = gelu(acc + bias)             (MLP-in)
// Block = 256 thr (8 waves); block tile 128x64; wave tile 32x32 (2x2 WMMA).
// Per K-step: A slice 128x32 (8KB) + B slice 64x32 (4KB) staged into LDS via
// GLOBAL_LOAD_ASYNC_TO_LDS_B128 (ASYNCcnt), ping-pong pipelined.
// ---------------------------------------------------------------------------
template <int MODE>
__global__ void gemm_bf16(const unsigned short* __restrict__ A,
                          const unsigned short* __restrict__ Bt,
                          const float* __restrict__ bias,
                          const float* __restrict__ resid,
                          float* __restrict__ outF,
                          unsigned short* __restrict__ outB,
                          int M, int N, int K) {
  __shared__ __align__(16) unsigned short As[2][128 * 32];
  __shared__ __align__(16) unsigned short Bs[2][64 * 32];

  int t    = threadIdx.x;
  int lane = t & 31;
  int wave = t >> 5;
  int half = lane >> 4, lr = lane & 15;
  int mw = (wave >> 1) * 32;          // wave tile origin within block tile
  int nw = (wave & 1) * 32;
  size_t m0b = (size_t)blockIdx.y * 128;
  size_t n0b = (size_t)blockIdx.x * 64;

  // Issue one K-slice worth of async copies (3 x 16B per thread).
  auto issue = [&](int buf, int k) {
#pragma unroll
    for (int i = 0; i < 2; ++i) {                      // A: 512 x 16B slots
      int slot = t + i * 256;
      int row = slot >> 2, ch = (slot & 3) * 8;
      const unsigned short* src = A + (m0b + row) * K + k + ch;
      async_cp16((unsigned)(uintptr_t)&As[buf][row * 32 + ch], src);
    }
    {                                                   // B: 256 x 16B slots
      int row = t >> 2, ch = (t & 3) * 8;
      const unsigned short* src = Bt + (n0b + row) * K + k + ch;
      async_cp16((unsigned)(uintptr_t)&Bs[buf][row * 32 + ch], src);
    }
  };

  v8f acc[2][2] = {};
  int nk = K >> 5;
  issue(0, 0);
  for (int kt = 0; kt < nk; ++kt) {
    int cur = kt & 1;
    if (kt + 1 < nk) {
      issue(cur ^ 1, (kt + 1) * 32);
      asm volatile("s_wait_asynccnt 3" ::: "memory");   // tile 'cur' landed
    } else {
      asm volatile("s_wait_asynccnt 0" ::: "memory");
    }
    __syncthreads();

    const unsigned short* Ab = &As[cur][0];
    const unsigned short* Bb = &Bs[cur][0];
    v16bf a0 = ld_a_frag(Ab + (mw + lr) * 32, half);
    v16bf a1 = ld_a_frag(Ab + (mw + 16 + lr) * 32, half);
    v16bf b0 = ld_b_frag(Bb + (nw + lr) * 32 + half * 16);
    v16bf b1 = ld_b_frag(Bb + (nw + 16 + lr) * 32 + half * 16);
    acc[0][0] = wmma_bf16(a0, b0, acc[0][0]);
    acc[0][1] = wmma_bf16(a0, b1, acc[0][1]);
    acc[1][0] = wmma_bf16(a1, b0, acc[1][0]);
    acc[1][1] = wmma_bf16(a1, b1, acc[1][1]);
    __syncthreads();                                    // done reading 'cur'
  }

  int m0 = (int)m0b + mw;
  int n0 = (int)n0b + nw;
#pragma unroll
  for (int mi = 0; mi < 2; mi++) {
#pragma unroll
    for (int ni = 0; ni < 2; ni++) {
      int col = n0 + ni * 16 + lr;
      float bv = bias[col];
#pragma unroll
      for (int r = 0; r < 8; r++) {
        int row = m0 + mi * 16 + r + half * 8;   // C layout: lanes0-15 M=r, lanes16-31 M=r+8
        float v = acc[mi][ni][r] + bv;
        if constexpr (MODE == 0) {
          int bq = row >> 10, s = row & 1023, h = col >> 6, e = col & 63;
          outB[(((size_t)bq * 12 + h) * 1024 + s) * 64 + e] = f2bf(v);
        } else if constexpr (MODE == 1) {
          int bq = row >> 10, s = row & 1023, h = col >> 6, e = col & 63;
          outB[(((size_t)bq * 12 + h) * 64 + e) * 1024 + s] = f2bf(v);
        } else if constexpr (MODE == 2) {
          size_t idx = (size_t)row * N + col;
          outF[idx] = resid[idx] + v;
        } else {
          outB[(size_t)row * N + col] = f2bf(gelu_new(v));
        }
      }
    }
  }
}

// ---------------------------------------------------------------------------
// Flash-style causal attention. One wave (32 thr) per 16 query rows.
//   Qb,Kb: [B,H,S,64] bf16.  Vt: [B,H,64,S] bf16.  Zb: [B*S, 768] bf16.
// Q*K^T and P*V via v_wmma_f32_16x16x32_bf16; online softmax in fp32 with
// shfl reductions over 16-lane halves; P transposed via an LDS tile.
// ---------------------------------------------------------------------------
__global__ void attn_kernel(const unsigned short* __restrict__ Qb,
                            const unsigned short* __restrict__ Kb,
                            const unsigned short* __restrict__ Vt,
                            unsigned short* __restrict__ Zb) {
  const int S = 1024, H = 12, Dh = 64;
  int lane = threadIdx.x;
  int half = lane >> 4, lr = lane & 15;
  int q0 = blockIdx.x * 16;
  int h  = blockIdx.y;
  int b  = blockIdx.z;

  size_t headOff = ((size_t)b * H + h) * S * Dh;
  const unsigned short* Qh = Qb + headOff;
  const unsigned short* Kh = Kb + headOff;
  const unsigned short* Vh = Vt + headOff;   // [Dh][S] within head

  v16bf aQ0 = ld_a_frag(Qh + (size_t)(q0 + lr) * Dh + 0, half);
  v16bf aQ1 = ld_a_frag(Qh + (size_t)(q0 + lr) * Dh + 32, half);

  v8f accz[4] = {};
  float mrow[8], lrow[8];
#pragma unroll
  for (int r = 0; r < 8; r++) { mrow[r] = -1e30f; lrow[r] = 0.0f; }

  __shared__ __align__(16) unsigned short ptile[16][32];

  int ntiles = (q0 >> 5) + 1;                // causal: keys <= q0+15
  for (int kt = 0; kt < ntiles; kt++) {
    int kb = kt * 32;
    v8f st[2];
#pragma unroll
    for (int nt = 0; nt < 2; nt++) {
      v8f s = {};
      int keycol = kb + nt * 16 + lr;
      const unsigned short* kr = Kh + (size_t)keycol * Dh;
      v16bf bk0 = ld_b_frag(kr + half * 16);
      v16bf bk1 = ld_b_frag(kr + 32 + half * 16);
      s = wmma_bf16(aQ0, bk0, s);
      s = wmma_bf16(aQ1, bk1, s);
#pragma unroll
      for (int r = 0; r < 8; r++) {
        int qrow = q0 + r + half * 8;
        s[r] = (keycol <= qrow) ? s[r] * 0.125f : -1e5f;   // scale 1/sqrt(64), mask
      }
      st[nt] = s;
    }

    float p0[8], p1[8];
#pragma unroll
    for (int r = 0; r < 8; r++) {
      float t = fmaxf(st[0][r], st[1][r]);
      t = fmaxf(t, __shfl_xor(t, 1));
      t = fmaxf(t, __shfl_xor(t, 2));
      t = fmaxf(t, __shfl_xor(t, 4));
      t = fmaxf(t, __shfl_xor(t, 8));        // row max within 16-lane half
      float mn = fmaxf(mrow[r], t);
      float sc = __expf(mrow[r] - mn);
      p0[r] = __expf(st[0][r] - mn);
      p1[r] = __expf(st[1][r] - mn);
      float ps = p0[r] + p1[r];
      ps += __shfl_xor(ps, 1);
      ps += __shfl_xor(ps, 2);
      ps += __shfl_xor(ps, 4);
      ps += __shfl_xor(ps, 8);
      lrow[r] = lrow[r] * sc + ps;
      mrow[r] = mn;
      accz[0][r] *= sc; accz[1][r] *= sc; accz[2][r] *= sc; accz[3][r] *= sc;
      // transpose P through LDS into A-fragment order
      ptile[r + half * 8][lr]      = f2bf(p0[r]);
      ptile[r + half * 8][lr + 16] = f2bf(p1[r]);
    }
    asm volatile("s_wait_dscnt 0" ::: "memory");
    v16bf aP = ld_a_frag(&ptile[lr][0], half);
#pragma unroll
    for (int t4 = 0; t4 < 4; t4++) {
      const unsigned short* vr = Vh + (size_t)(t4 * 16 + lr) * S + kb + half * 16;
      v16bf bv = ld_b_frag(vr);
      accz[t4] = wmma_bf16(aP, bv, accz[t4]);
    }
    asm volatile("s_wait_dscnt 0" ::: "memory");
  }

#pragma unroll
  for (int t4 = 0; t4 < 4; t4++) {
#pragma unroll
    for (int r = 0; r < 8; r++) {
      int row = b * S + q0 + r + half * 8;
      int col = h * 64 + t4 * 16 + lr;
      Zb[(size_t)row * 768 + col] = f2bf(accz[t4][r] / lrow[r]);
    }
  }
}

// ---------------------------------------------------------------------------
// Launch
// ---------------------------------------------------------------------------
extern "C" void kernel_launch(void* const* d_in, const int* in_sizes, int n_in,
                              void* d_out, int out_size, void* d_ws, size_t ws_size,
                              hipStream_t stream) {
  (void)in_sizes; (void)n_in; (void)out_size; (void)ws_size;
  const float* resid_pre = (const float*)d_in[0];
  const float* ln1_w = (const float*)d_in[1];
  const float* ln1_b = (const float*)d_in[2];
  const float* W_Q   = (const float*)d_in[3];
  const float* W_K   = (const float*)d_in[4];
  const float* W_V   = (const float*)d_in[5];
  const float* W_O   = (const float*)d_in[6];
  const float* b_Q   = (const float*)d_in[7];
  const float* b_K   = (const float*)d_in[8];
  const float* b_V   = (const float*)d_in[9];
  const float* b_O   = (const float*)d_in[10];
  const float* ln2_w = (const float*)d_in[11];
  const float* ln2_b = (const float*)d_in[12];
  const float* W_in  = (const float*)d_in[13];
  const float* b_in  = (const float*)d_in[14];
  const float* W_out = (const float*)d_in[15];
  const float* b_out = (const float*)d_in[16];

  char* ws = (char*)d_ws;
  // weight slabs (bf16)
  unsigned short* WQt   = (unsigned short*)(ws + 0);          // 768*768*2
  unsigned short* WKt   = (unsigned short*)(ws + 1179648);
  unsigned short* WVt   = (unsigned short*)(ws + 2359296);
  unsigned short* WOt   = (unsigned short*)(ws + 3538944);
  unsigned short* WinT  = (unsigned short*)(ws + 4718592);    // 3072*768*2
  unsigned short* WoutT = (unsigned short*)(ws + 9437184);    // 768*3072*2
  // activation region A (25165824 B): ln1b,Qb,Kb,Vt -> later reused as hb
  unsigned short* ln1b = (unsigned short*)(ws + 14155776);    // 4096*768*2
  unsigned short* Qb   = (unsigned short*)(ws + 20447232);
  unsigned short* Kb   = (unsigned short*)(ws + 26738688);
  unsigned short* Vtb  = (unsigned short*)(ws + 33030144);
  unsigned short* hb   = (unsigned short*)(ws + 14155776);    // reuse (25.2 MB)
  unsigned short* Zb   = (unsigned short*)(ws + 39321600);    // 4096*768*2
  unsigned short* ln2b = (unsigned short*)(ws + 39321600);    // reuse after O-proj
  float* resid_mid     = (float*)(ws + 45613056);             // 4096*768*4

  // 1) pack weights to bf16 (transposed [N][K])
  pack_qkv_w<<<2304, 256, 0, stream>>>(W_Q, WQt);
  pack_qkv_w<<<2304, 256, 0, stream>>>(W_K, WKt);
  pack_qkv_w<<<2304, 256, 0, stream>>>(W_V, WVt);
  transpose_pack<<<(768 * 768 + 255) / 256, 256, 0, stream>>>(W_O, WOt, 768, 768);
  transpose_pack<<<(768 * 3072 + 255) / 256, 256, 0, stream>>>(W_in, WinT, 768, 3072);
  transpose_pack<<<(3072 * 768 + 255) / 256, 256, 0, stream>>>(W_out, WoutT, 3072, 768);

  // 2) LN1
  ln_kernel<<<4096, 256, 0, stream>>>(resid_pre, ln1_w, ln1_b, ln1b);

  // 3) QKV projections (M=4096, N=768, K=768)
  dim3 g768(12, 32), blk(256);
  gemm_bf16<0><<<g768, blk, 0, stream>>>(ln1b, WQt, b_Q, nullptr, nullptr, Qb, 4096, 768, 768);
  gemm_bf16<0><<<g768, blk, 0, stream>>>(ln1b, WKt, b_K, nullptr, nullptr, Kb, 4096, 768, 768);
  gemm_bf16<1><<<g768, blk, 0, stream>>>(ln1b, WVt, b_V, nullptr, nullptr, Vtb, 4096, 768, 768);

  // 4) attention
  attn_kernel<<<dim3(64, 12, 4), 32, 0, stream>>>(Qb, Kb, Vtb, Zb);

  // 5) O-projection + residual -> resid_mid (fp32)
  gemm_bf16<2><<<g768, blk, 0, stream>>>(Zb, WOt, b_O, resid_pre, resid_mid, nullptr, 4096, 768, 768);

  // 6) LN2
  ln_kernel<<<4096, 256, 0, stream>>>(resid_mid, ln2_w, ln2_b, ln2b);

  // 7) MLP in + GELU (N=3072)
  gemm_bf16<3><<<dim3(48, 32), blk, 0, stream>>>(ln2b, WinT, b_in, nullptr, nullptr, hb, 4096, 3072, 768);

  // 8) MLP out + residual -> d_out (K=3072)
  gemm_bf16<2><<<g768, blk, 0, stream>>>(hb, WoutT, b_out, resid_mid, (float*)d_out, nullptr, 4096, 768, 3072);
}
